// WindowAttention_48232482734195
// MI455X (gfx1250) — matrix operand order repaired and checked by reading, not compiled, not run on hardware
//
#include <hip/hip_runtime.h>

typedef __attribute__((ext_vector_type(16))) _Float16 v16h;
typedef __attribute__((ext_vector_type(8)))  _Float16 v8h;
typedef __attribute__((ext_vector_type(4)))  _Float16 v4h;
typedef __attribute__((ext_vector_type(8)))  float    v8f;
typedef __attribute__((ext_vector_type(4)))  float    v4f;

__device__ __forceinline__ v16h load16(const _Float16* p0, const _Float16* p1) {
    union { v16h v; v8h h[2]; } u;
    u.h[0] = *(const v8h*)p0;
    u.h[1] = *(const v8h*)p1;
    return u.v;
}

__device__ __forceinline__ v8h pack8(v4f a, v4f b) {
    v8h r;
    #pragma unroll
    for (int i = 0; i < 4; ++i) { r[i] = (_Float16)a[i]; r[i + 4] = (_Float16)b[i]; }
    return r;
}

__device__ __forceinline__ v8f wmma_f16(v16h a, v16h b, v8f c) {
    return __builtin_amdgcn_wmma_f32_16x16x32_f16(false, a, false, b, (short)0, c, false, false);
}

// ---------------------------------------------------------------------------
// Precompute fused (rel-pos bias + window mask) table, padded to 64x64.
//   n>=49        -> -1e30  (raw padded score is 0, so 0 + (-1e30) kills col)
//   m>=49, n<49  -> 0      (padded query rows: harmless uniform softmax)
//   else         -> rpb[rel_idx[m][n]][h] + mask[wm][m][n]
// ---------------------------------------------------------------------------
__global__ __launch_bounds__(256)
void build_bias_tab(const float* __restrict__ mask, const float* __restrict__ rpb,
                    const int* __restrict__ rel_idx, float* __restrict__ Btab) {
    const int wm = blockIdx.x / 12;
    const int h  = blockIdx.x % 12;
    float* out = Btab + (size_t)blockIdx.x * 64 * 64;
    for (int e = threadIdx.x; e < 64 * 64; e += 256) {
        int m = e >> 6, n = e & 63;
        float v;
        if (n >= 49)      v = -1e30f;
        else if (m >= 49) v = 0.0f;
        else              v = rpb[rel_idx[m * 49 + n] * 12 + h]
                            + mask[(size_t)wm * 49 * 49 + m * 49 + n];
        out[e] = v;
    }
}

// ---------------------------------------------------------------------------
// Generic Y = (X @ W^T + bias) * (ncol < oscale_cols ? oscale : 1).
// X: MxK (f32 or f16), W: NoutxK (f32), Y: MxNout (f16 or f32).
// Tile: 64 rows x 128 cols per WG (256 thr = 8 waves); K stepped by 32.
// Double-buffered LDS: prefetch tile k+1 into registers while computing k.
// Requires M%64==0, Nout%128==0, K%32==0.
// ---------------------------------------------------------------------------
template<bool XF16, bool YF32>
__global__ __launch_bounds__(256)
void gemm_xwT(const void* __restrict__ Xv, const float* __restrict__ W,
              const float* __restrict__ bias, int K, int Nout,
              void* __restrict__ Yv, float oscale, int oscale_cols) {
    const int m0  = blockIdx.x * 64;
    const int n0  = blockIdx.y * 128;
    const int tid = threadIdx.x;
    const int wv  = tid >> 5;
    const int lane = tid & 31;
    const int ml  = lane & 15;
    const int hb  = lane >> 4;
    const int mi  = wv & 3;              // 16-row subtile
    const int nq  = (wv >> 2) * 4;       // first 16-col subtile

    __shared__ __align__(16) _Float16 Xs[2][64 * 40];
    __shared__ __align__(16) _Float16 Ws[2][128 * 40];

    // fixed per-thread staging slots
    const int xr = tid >> 2;             // X row 0..63
    const int xc = (tid & 3) * 8;        // X cols {0,8,16,24}
    const int wr = tid >> 1;             // W row 0..127
    const int wc = (tid & 1) * 16;       // W cols {0,16}

    const float*    Xf = (const float*)Xv;
    const _Float16* Xh = (const _Float16*)Xv;

    v4f xa, xb; v8h xh;                  // X prefetch regs
    v4f wrg[4];                          // W prefetch regs

    auto fetch = [&](int kk) {
        if (XF16) {
            xh = *(const v8h*)&Xh[(size_t)(m0 + xr) * K + kk + xc];
        } else {
            const float* xp = &Xf[(size_t)(m0 + xr) * K + kk + xc];
            xa = *(const v4f*)xp;
            xb = *(const v4f*)(xp + 4);
        }
        const float* wp = &W[(size_t)(n0 + wr) * K + kk + wc];
        wrg[0] = *(const v4f*)wp;
        wrg[1] = *(const v4f*)(wp + 4);
        wrg[2] = *(const v4f*)(wp + 8);
        wrg[3] = *(const v4f*)(wp + 12);
    };
    auto stage = [&](int buf) {
        *(v8h*)&Xs[buf][xr * 40 + xc]     = XF16 ? xh : pack8(xa, xb);
        *(v8h*)&Ws[buf][wr * 40 + wc]     = pack8(wrg[0], wrg[1]);
        *(v8h*)&Ws[buf][wr * 40 + wc + 8] = pack8(wrg[2], wrg[3]);
    };

    fetch(0);
    stage(0);
    __syncthreads();

    v8f acc[4] = {};
    const int nk = K / 32;
    for (int ki = 0; ki < nk; ++ki) {
        const int cur = ki & 1;
        if (ki + 1 < nk) fetch((ki + 1) * 32);

        const _Float16* xsb = Xs[cur];
        const _Float16* wsb = Ws[cur];
        const int arow = mi * 16 + ml;
        v16h a = load16(&xsb[arow * 40 + hb * 8], &xsb[arow * 40 + hb * 8 + 16]);
        const int bbase = hb * 8;
        #pragma unroll
        for (int t = 0; t < 4; ++t) {
            int brow = (nq + t) * 16 + ml;
            v16h b = load16(&wsb[brow * 40 + bbase], &wsb[brow * 40 + bbase + 16]);
            acc[t] = wmma_f16(a, b, acc[t]);
        }
        if (ki + 1 < nk) stage(cur ^ 1);
        __syncthreads();
    }

    #pragma unroll
    for (int t = 0; t < 4; ++t) {
        int ncol = n0 + (nq + t) * 16 + ml;
        float bv = bias[ncol];
        float f  = (ncol < oscale_cols) ? oscale : 1.0f;
        #pragma unroll
        for (int j = 0; j < 8; ++j) {
            int mrow = m0 + mi * 16 + hb * 8 + j;
            float v = (acc[t][j] + bv) * f;
            if (YF32) ((float*)Yv)[(size_t)mrow * Nout + ncol] = v;
            else      ((_Float16*)Yv)[(size_t)mrow * Nout + ncol] = (_Float16)v;
        }
    }
}

// ---------------------------------------------------------------------------
// Attention: one WG (128 thr = 4 waves) per (window, head).  N=49 padded to 64.
// S = q@k^T + Btab ; softmax ; O_se = P@v_se ; O_de = P@v_de.
// (q already carries the 1/sqrt(hd) scale from the qkv GEMM epilogue.)
// ---------------------------------------------------------------------------
__global__ __launch_bounds__(128)
void attn_kernel(const _Float16* __restrict__ qkv,   // (BW*49, 1152) f16
                 const _Float16* __restrict__ vse,   // (BW*49, 384)  f16
                 const float* __restrict__ Btab,     // (4,12,64,64)  f32
                 _Float16* __restrict__ o_se,        // (BW*49, 384)  f16
                 _Float16* __restrict__ o_de) {
    const int w = blockIdx.x;
    const int h = blockIdx.y;
    const int tid = threadIdx.x;
    const int wv = tid >> 5, lane = tid & 31;
    const int ml = lane & 15, hb = lane >> 4;

    __shared__ __align__(16) _Float16 Qs[64 * 40];
    __shared__ __align__(16) _Float16 Ks[64 * 40];
    __shared__ __align__(16) _Float16 VseT[32 * 72];  // transposed: [d][n]
    __shared__ __align__(16) _Float16 VdeT[32 * 72];
    __shared__ __align__(16) float S[64 * 68];        // 272B row stride (16B-mult)
    __shared__ __align__(16) _Float16 P[64 * 72];

    const size_t rowbase = (size_t)w * 49;

    // stage Q, K (row-major) and V_se^T, V_de^T; zero-pad rows 49..63.
    for (int idx = tid; idx < 64 * 4; idx += 128) {
        int n = idx >> 2, c = idx & 3;
        v8h qv = {}, kv = {}, vs = {}, vd = {};
        if (n < 49) {
            const _Float16* src = qkv + (rowbase + n) * 1152 + h * 32 + c * 8;
            qv = *(const v8h*)(src);
            kv = *(const v8h*)(src + 384);
            vd = *(const v8h*)(src + 768);
            vs = *(const v8h*)(vse + (rowbase + n) * 384 + h * 32 + c * 8);
        }
        *(v8h*)&Qs[n * 40 + c * 8] = qv;
        *(v8h*)&Ks[n * 40 + c * 8] = kv;
        #pragma unroll
        for (int i = 0; i < 8; ++i) {
            int d = c * 8 + i;
            VseT[d * 72 + n] = vs[i];
            VdeT[d * 72 + n] = vd[i];
        }
    }
    __syncthreads();

    // ---- scores: each wave computes rows [m0, m0+16) over all 64 cols -----
    const int m0 = wv * 16;
    {
        const int arow = m0 + ml;
        v16h a = load16(&Qs[arow * 40 + hb * 8], &Qs[arow * 40 + hb * 8 + 16]);
        const int bbase = hb * 8;
        v8f c[4];
        #pragma unroll
        for (int nt = 0; nt < 4; ++nt) {
            int brow = nt * 16 + ml;             // B(kk,n) = K[n][kk], table order
            v16h b = load16(&Ks[brow * 40 + bbase], &Ks[brow * 40 + bbase + 16]);
            v8f z = {};
            c[nt] = wmma_f16(a, b, z);
        }
        const float* bt = Btab + (size_t)((w & 3) * 12 + h) * 64 * 64;
        #pragma unroll
        for (int nt = 0; nt < 4; ++nt) {
            int n = nt * 16 + ml;
            #pragma unroll
            for (int j = 0; j < 8; ++j) {
                int m = m0 + hb * 8 + j;
                S[m * 68 + n] = c[nt][j] + bt[m * 64 + n];
            }
        }
    }
    __syncthreads();

    // ---- softmax: one thread per row, vectorized v4f over 64 cols ----
    if (tid < 64) {
        float* Sr = &S[tid * 68];
        float mx = -1e30f;
        #pragma unroll
        for (int i = 0; i < 16; ++i) {
            v4f x = *(const v4f*)&Sr[i * 4];
            mx = fmaxf(mx, fmaxf(fmaxf(x[0], x[1]), fmaxf(x[2], x[3])));
        }
        float sum = 0.f;
        #pragma unroll
        for (int i = 0; i < 16; ++i) {
            v4f x = *(v4f*)&Sr[i * 4];
            v4f e;
            e[0] = __expf(x[0] - mx); e[1] = __expf(x[1] - mx);
            e[2] = __expf(x[2] - mx); e[3] = __expf(x[3] - mx);
            sum += (e[0] + e[1]) + (e[2] + e[3]);
            *(v4f*)&Sr[i * 4] = e;
        }
        float rs = 1.0f / sum;
        #pragma unroll
        for (int i = 0; i < 16; ++i) {
            v4f x = *(const v4f*)&Sr[i * 4];
            v4h p;
            p[0] = (_Float16)(x[0] * rs); p[1] = (_Float16)(x[1] * rs);
            p[2] = (_Float16)(x[2] * rs); p[3] = (_Float16)(x[3] * rs);
            *(v4h*)&P[tid * 72 + i * 4] = p;
        }
    }
    __syncthreads();

    // ---- O = P @ V : K=64 (2 steps of 32), N=32 (2 subtiles) ----
    v8f ose[2] = {}, ode[2] = {};
    #pragma unroll
    for (int kc = 0; kc < 2; ++kc) {
        const int arow = m0 + ml;
        const int abase = kc * 32 + hb * 8;
        v16h a = load16(&P[arow * 72 + abase], &P[arow * 72 + abase + 16]);
        const int bbase = kc * 32 + hb * 8;      // table order in VT rows
        #pragma unroll
        for (int dt = 0; dt < 2; ++dt) {
            int brow = dt * 16 + ml;
            v16h bs = load16(&VseT[brow * 72 + bbase], &VseT[brow * 72 + bbase + 16]);
            ose[dt] = wmma_f16(a, bs, ose[dt]);
            v16h bd = load16(&VdeT[brow * 72 + bbase], &VdeT[brow * 72 + bbase + 16]);
            ode[dt] = wmma_f16(a, bd, ode[dt]);
        }
    }
    #pragma unroll
    for (int dt = 0; dt < 2; ++dt) {
        #pragma unroll
        for (int j = 0; j < 8; ++j) {
            int m = m0 + hb * 8 + j;
            if (m < 49) {
                int d = dt * 16 + ml;
                size_t off = (rowbase + m) * 384 + h * 32 + d;
                o_se[off] = (_Float16)ose[dt][j];
                o_de[off] = (_Float16)ode[dt][j];
            }
        }
    }
}

// ---------------------------------------------------------------------------
extern "C" void kernel_launch(void* const* d_in, const int* in_sizes, int n_in,
                              void* d_out, int out_size, void* d_ws, size_t ws_size,
                              hipStream_t stream) {
    const float* se        = (const float*)d_in[0];
    const float* de        = (const float*)d_in[1];
    const float* mask      = (const float*)d_in[2];
    const float* w_v_se    = (const float*)d_in[3];
    const float* b_v_se    = (const float*)d_in[4];
    const float* w_qkv     = (const float*)d_in[5];
    const float* b_qkv     = (const float*)d_in[6];
    const float* w_proj_se = (const float*)d_in[7];
    const float* b_proj_se = (const float*)d_in[8];
    const float* w_proj_de = (const float*)d_in[9];
    const float* b_proj_de = (const float*)d_in[10];
    const float* rpb       = (const float*)d_in[11];
    const int*   rel_idx   = (const int*)d_in[12];

    const int M = 2048 * 49;                       // 100352 token rows
    const float SCALE = 0.17677669529663687f;      // 1/sqrt(32)

    _Float16* qkv_ws = (_Float16*)d_ws;            // M x 1152
    _Float16* vse_ws = qkv_ws + (size_t)M * 1152;  // M x 384
    _Float16* ose_ws = vse_ws + (size_t)M * 384;   // M x 384
    _Float16* ode_ws = ose_ws + (size_t)M * 384;   // M x 384
    float*    Btab   = (float*)(ode_ws + (size_t)M * 384); // 4*12*64*64 f32

    // 0) fused bias table (tiny)
    build_bias_tab<<<48, 256, 0, stream>>>(mask, rpb, rel_idx, Btab);

    // 1) input projections (f32 -> f16); q cols (0..383) get the 1/sqrt(hd) scale
    gemm_xwT<false, false><<<dim3(M / 64, 1152 / 128), 256, 0, stream>>>(
        de, w_qkv, b_qkv, 384, 1152, qkv_ws, SCALE, 384);
    gemm_xwT<false, false><<<dim3(M / 64, 384 / 128), 256, 0, stream>>>(
        se, w_v_se, b_v_se, 384, 384, vse_ws, 1.0f, 0);

    // 2) windowed attention, one WG per (window, head)
    attn_kernel<<<dim3(2048, 12), 128, 0, stream>>>(
        qkv_ws, vse_ws, Btab, ose_ws, ode_ws);

    // 3) output projections (f16 -> f32) straight into d_out
    float* out_se = (float*)d_out;
    float* out_de = out_se + (size_t)M * 384;
    gemm_xwT<true, true><<<dim3(M / 64, 384 / 128), 256, 0, stream>>>(
        ose_ws, w_proj_se, b_proj_se, 384, 384, out_se, 1.0f, 0);
    gemm_xwT<true, true><<<dim3(M / 64, 384 / 128), 256, 0, stream>>>(
        ode_ws, w_proj_de, b_proj_de, 384, 384, out_de, 1.0f, 0);
}
